// GraphEncoder_68264210202826
// MI455X (gfx1250) — matrix operand order repaired and checked by reading, not compiled, not run on hardware
//
#include <hip/hip_runtime.h>

typedef __attribute__((ext_vector_type(16))) _Float16 v16h;
typedef __attribute__((ext_vector_type(8)))  float    v8f;

#define HDIM   64
#define H2DIM  128
#define OUTC   128
#define LN_EPS 1e-5f
#define MSG_EPS 1e-7f
#define ENC_NEGINF 0x007FFFFFu   // encf(-inf)

// ---------- monotonic float<->uint encoding for atomicMax on floats ----------
__device__ __forceinline__ unsigned encf(float f) {
  unsigned u = __float_as_uint(f);
  return (u & 0x80000000u) ? ~u : (u | 0x80000000u);
}
__device__ __forceinline__ float decf(unsigned e) {
  return (e & 0x80000000u) ? __uint_as_float(e ^ 0x80000000u) : __uint_as_float(~e);
}

// ---------- WMMA fragment helpers (CDNA5 v_wmma_f32_16x16x32_f16) ----------
union FragH { v16h v; float4 f4[2]; _Float16 h[16]; };

__device__ __forceinline__ v8f wmma32(v16h a, v16h b, v8f c) {
  return __builtin_amdgcn_wmma_f32_16x16x32_f16(false, a, false, b, (short)0, c, false, false);
}

// A: 16x32 tile of row-major [*, ld] f16 matrix at (base row, kbase).
// lanes 0-15: row=lane, K = 0..7 & 16..23 ; lanes 16-31: row=lane-16, K = 8..15 & 24..31
// => per lane: two contiguous 16B chunks.
__device__ __forceinline__ v16h load_frag_a(const _Float16* A, int ld, int lane, int kbase) {
  int r = lane & 15, hi = lane >> 4;
  const _Float16* row = A + r * ld + kbase + hi * 8;
  FragH f;
  f.f4[0] = *(const float4*)(row);        // K offsets 0..7
  f.f4[1] = *(const float4*)(row + 16);   // K offsets 16..23
  return f.v;
}

// B fragments pre-packed: record per (frag, lane) = 16 contiguous halves (32B).
__device__ __forceinline__ v16h load_frag_bp(const _Float16* P, int frag, int lane) {
  const float4* p = (const float4*)(P + ((size_t)frag * 32 + lane) * 16);
  FragH f;
  f.f4[0] = p[0];
  f.f4[1] = p[1];
  return f.v;
}

// ---------- small utility kernels ----------
__global__ void k_fill_u32(unsigned* p, unsigned val, int n) {
  int i = blockIdx.x * blockDim.x + threadIdx.x;
  if (i < n) p[i] = val;
}

// Pack row-major f32 weight [K, Nc] into WMMA-B fragment records (f16).
// frag = kchunk * (Nc/16) + ntile ; within frag: lane-major, 16 halves each.
// lane: col n=(lane&15)+ntile*16, K = (lane>>4)*8 + 0..7 and 16 + (lane>>4)*8 + 0..7 within the 32-K chunk.
__global__ void k_pack_b(const float* __restrict__ W, _Float16* __restrict__ P,
                         int K, int Nc) {
  int i = blockIdx.x * blockDim.x + threadIdx.x;
  if (i >= K * Nc) return;
  int frag = i >> 9, idx = i & 511;
  int lane = idx >> 4, hh = idx & 15;
  int ntiles = Nc >> 4;
  int kchunk = frag / ntiles, ntile = frag - kchunk * ntiles;
  int n = (lane & 15) + ntile * 16;
  int k = kchunk * 32 + (lane >> 4) * 8 + (hh & 7) + (hh >> 3) * 16;
  P[i] = (_Float16)W[k * Nc + n];
}

// ---------- edge pass 1: per-(dst, channel) max of s = (relu(x_src+ea)+eps)*t ----------
__global__ void k_edge_max(const float* __restrict__ xin, const float* __restrict__ ea,
                           const int* __restrict__ src, const int* __restrict__ dst,
                           const float* __restrict__ tptr, int layer,
                           unsigned* __restrict__ smax, int E) {
  int gid = blockIdx.x * blockDim.x + threadIdx.x;
  int e = gid >> 4;
  if (e >= E) return;
  int c = (gid & 15) * 4;
  float t = tptr[layer];
  int s = src[e], d = dst[e];
  float4 xv = *(const float4*)(xin + s * HDIM + c);
  float4 av = *(const float4*)(ea + e * HDIM + c);
  unsigned* base = smax + d * HDIM + c;
  atomicMax(base + 0, encf((fmaxf(xv.x + av.x, 0.f) + MSG_EPS) * t));
  atomicMax(base + 1, encf((fmaxf(xv.y + av.y, 0.f) + MSG_EPS) * t));
  atomicMax(base + 2, encf((fmaxf(xv.z + av.z, 0.f) + MSG_EPS) * t));
  atomicMax(base + 3, encf((fmaxf(xv.w + av.w, 0.f) + MSG_EPS) * t));
}

// ---------- edge pass 2: accumulate num = sum(msg*e), den = sum(e) ----------
__global__ void k_edge_accum(const float* __restrict__ xin, const float* __restrict__ ea,
                             const int* __restrict__ src, const int* __restrict__ dst,
                             const float* __restrict__ tptr, int layer,
                             const unsigned* __restrict__ smax,
                             float* __restrict__ num, float* __restrict__ den, int E) {
  int gid = blockIdx.x * blockDim.x + threadIdx.x;
  int e = gid >> 4;
  if (e >= E) return;
  int c = (gid & 15) * 4;
  float t = tptr[layer];
  int s = src[e], d = dst[e];
  float4 xv = *(const float4*)(xin + s * HDIM + c);
  float4 av = *(const float4*)(ea + e * HDIM + c);
  float msg[4] = { fmaxf(xv.x + av.x, 0.f) + MSG_EPS, fmaxf(xv.y + av.y, 0.f) + MSG_EPS,
                   fmaxf(xv.z + av.z, 0.f) + MSG_EPS, fmaxf(xv.w + av.w, 0.f) + MSG_EPS };
  int nb = d * HDIM + c;
#pragma unroll
  for (int j = 0; j < 4; ++j) {
    float mx = decf(smax[nb + j]);
    if (!(mx > -3.0e38f)) mx = 0.f;   // empty-segment fixup (matches reference)
    float ev = __expf(msg[j] * t - mx);
    atomicAdd(den + nb + j, ev);
    atomicAdd(num + nb + j, msg[j] * ev);
  }
}

// ---------- combine: a16 = f16( num/(den+1e-16) + xin ), 4 elems/thread ----------
__global__ void k_combine(const float* __restrict__ num, const float* __restrict__ den,
                          const float* __restrict__ xin, _Float16* __restrict__ a16, int n4) {
  int i = blockIdx.x * blockDim.x + threadIdx.x;
  if (i >= n4) return;
  float4 nu = *(const float4*)(num + i * 4);
  float4 de = *(const float4*)(den + i * 4);
  float4 xv = *(const float4*)(xin + i * 4);
  union { _Float16 h[4]; unsigned long long u; } o;
  o.h[0] = (_Float16)(nu.x / (de.x + 1e-16f) + xv.x);
  o.h[1] = (_Float16)(nu.y / (de.y + 1e-16f) + xv.y);
  o.h[2] = (_Float16)(nu.z / (de.z + 1e-16f) + xv.z);
  o.h[3] = (_Float16)(nu.w / (de.w + 1e-16f) + xv.w);
  *(unsigned long long*)(a16 + i * 4) = o.u;
}

// ---------- GEMM1 [N,64]x[64,128] + bias + LayerNorm + ReLU -> f16 ----------
__global__ void k_gemm1(const _Float16* __restrict__ A, const _Float16* __restrict__ Wp,
                        const float* __restrict__ bias, const float* __restrict__ lg,
                        const float* __restrict__ lb, _Float16* __restrict__ out16) {
  __shared__ float tile[16][H2DIM + 1];
  __shared__ float mu[16], rs[16];
  int row0 = blockIdx.x * 16;
  int wave = threadIdx.x >> 5, lane = threadIdx.x & 31;
  v8f c = {};
#pragma unroll
  for (int kc = 0; kc < HDIM; kc += 32) {
    v16h a = load_frag_a(A + row0 * HDIM, HDIM, lane, kc);
    v16h b = load_frag_bp(Wp, (kc >> 5) * 8 + wave, lane);
    c = wmma32(a, b, c);
  }
  int n = (lane & 15) + wave * 16, hi = lane >> 4;
#pragma unroll
  for (int v = 0; v < 8; ++v) tile[v + hi * 8][n] = c[v] + bias[n];
  __syncthreads();
  if (threadIdx.x < 16) {
    float s = 0.f, s2 = 0.f;
    for (int j = 0; j < H2DIM; ++j) { float x = tile[threadIdx.x][j]; s += x; s2 += x * x; }
    float m = s * (1.f / H2DIM);
    mu[threadIdx.x] = m;
    rs[threadIdx.x] = rsqrtf(s2 * (1.f / H2DIM) - m * m + LN_EPS);
  }
  __syncthreads();
  // each thread: 8 contiguous columns of one row -> one 16B f16 store
  {
    int r = threadIdx.x >> 4, c0 = (threadIdx.x & 15) * 8;
    float m = mu[r], rr = rs[r];
    union { _Float16 h[8]; float4 f4; } o;
#pragma unroll
    for (int j = 0; j < 8; ++j) {
      int cc = c0 + j;
      float x = (tile[r][cc] - m) * rr * lg[cc] + lb[cc];
      o.h[j] = (_Float16)fmaxf(x, 0.f);
    }
    *(float4*)(out16 + (row0 + r) * H2DIM + c0) = o.f4;
  }
}

// ---------- GEMM2 [N,128]x[128,64] + bias, store or residual-add into X ----------
__global__ void k_gemm2(const _Float16* __restrict__ A, const _Float16* __restrict__ Wp,
                        const float* __restrict__ bias, float* __restrict__ X, int addmode) {
  int row0 = blockIdx.x * 16;
  int wave = threadIdx.x >> 5, lane = threadIdx.x & 31;
  v8f c = {};
#pragma unroll
  for (int kc = 0; kc < H2DIM; kc += 32) {
    v16h a = load_frag_a(A + row0 * H2DIM, H2DIM, lane, kc);
    v16h b = load_frag_bp(Wp, (kc >> 5) * 4 + wave, lane);
    c = wmma32(a, b, c);
  }
  int n = (lane & 15) + wave * 16, hi = lane >> 4;
#pragma unroll
  for (int v = 0; v < 8; ++v) {
    int r = row0 + v + hi * 8;
    float val = c[v] + bias[n];
    if (addmode) X[r * HDIM + n] += val; else X[r * HDIM + n] = val;
  }
}

// ---------- final GEMM [N,64]x[64,128] + bias -> f32 out ----------
__global__ void k_gemm_out(const _Float16* __restrict__ A, const _Float16* __restrict__ Wp,
                           const float* __restrict__ bias, float* __restrict__ out) {
  int row0 = blockIdx.x * 16;
  int wave = threadIdx.x >> 5, lane = threadIdx.x & 31;
  v8f c = {};
#pragma unroll
  for (int kc = 0; kc < HDIM; kc += 32) {
    v16h a = load_frag_a(A + row0 * HDIM, HDIM, lane, kc);
    v16h b = load_frag_bp(Wp, (kc >> 5) * 8 + wave, lane);
    c = wmma32(a, b, c);
  }
  int n = (lane & 15) + wave * 16, hi = lane >> 4;
#pragma unroll
  for (int v = 0; v < 8; ++v)
    out[(row0 + v + hi * 8) * OUTC + n] = c[v] + bias[n];
}

// ---------- wave-per-node LayerNorm + mish ----------
__device__ __forceinline__ float mishf(float x) {
  float sp = (x > 20.f) ? x : log1pf(__expf(x));
  return x * tanhf(sp);
}

__global__ void k_prenorm_mish(const float* __restrict__ X, const float* __restrict__ g,
                               const float* __restrict__ b, float* __restrict__ hout,
                               _Float16* __restrict__ hout16, int writeHalf, int N) {
  int node = blockIdx.x * 8 + (threadIdx.x >> 5);
  if (node >= N) return;
  int lane = threadIdx.x & 31;
  float2 v = *(const float2*)(X + node * HDIM + lane * 2);
  float s = v.x + v.y, s2 = v.x * v.x + v.y * v.y;
#pragma unroll
  for (int off = 16; off > 0; off >>= 1) {
    s  += __shfl_xor(s,  off, 32);
    s2 += __shfl_xor(s2, off, 32);
  }
  float mu = s * (1.f / HDIM);
  float r = rsqrtf(s2 * (1.f / HDIM) - mu * mu + LN_EPS);
  int c0 = lane * 2;
  float m0 = mishf((v.x - mu) * r * g[c0] + b[c0]);
  float m1 = mishf((v.y - mu) * r * g[c0 + 1] + b[c0 + 1]);
  *(float2*)(hout + node * HDIM + c0) = make_float2(m0, m1);
  if (writeHalf) {
    union { _Float16 h[2]; unsigned u; } o;
    o.h[0] = (_Float16)m0;
    o.h[1] = (_Float16)m1;
    *(unsigned*)(hout16 + node * HDIM + c0) = o.u;
  }
}

// ---------- mean-pool accumulation ----------
__global__ void k_pool(const float* __restrict__ y, const int* __restrict__ batch,
                       float* __restrict__ gsum, float* __restrict__ gcnt, int N) {
  int gid = blockIdx.x * blockDim.x + threadIdx.x;
  int node = gid >> 4;
  if (node >= N) return;
  int c = (gid & 15) * 4;
  int bg = batch[node];
  float4 v = *(const float4*)(y + node * HDIM + c);
  float* base = gsum + bg * HDIM + c;
  atomicAdd(base + 0, v.x);
  atomicAdd(base + 1, v.y);
  atomicAdd(base + 2, v.z);
  atomicAdd(base + 3, v.w);
  if (c == 0) atomicAdd(gcnt + bg, 1.f);
}

// ---------- pooled GEMM (G=64 rows, trivial) ----------
__global__ void k_pool_out(const float* __restrict__ gsum, const float* __restrict__ gcnt,
                           const float* __restrict__ W, const float* __restrict__ bias,
                           float* __restrict__ out) {
  int g = blockIdx.x, o = threadIdx.x;
  float inv = 1.f / fmaxf(gcnt[g], 1.f);
  float acc = bias[o];
  for (int c = 0; c < HDIM; ++c) acc += gsum[g * HDIM + c] * inv * W[c * OUTC + o];
  out[g * OUTC + o] = acc;
}

// =======================================================================
extern "C" void kernel_launch(void* const* d_in, const int* in_sizes, int n_in,
                              void* d_out, int out_size, void* d_ws, size_t ws_size,
                              hipStream_t stream) {
  const float* x     = (const float*)d_in[0];
  const int*   eidx  = (const int*)d_in[1];     // [2,E]: row0 = src, row1 = dst
  const float* ea    = (const float*)d_in[2];
  const int*   batch = (const int*)d_in[3];
  const float* t     = (const float*)d_in[4];
  const float* W1    = (const float*)d_in[5];   // [L,64,128]
  const float* b1    = (const float*)d_in[6];   // [L,128]
  const float* lng   = (const float*)d_in[7];
  const float* lnb   = (const float*)d_in[8];
  const float* W2    = (const float*)d_in[9];   // [L,128,64]
  const float* b2    = (const float*)d_in[10];  // [L,64]
  const float* ng    = (const float*)d_in[11];  // [L,64]
  const float* nb    = (const float*)d_in[12];
  const float* linW  = (const float*)d_in[13];  // [64,128]
  const float* linb  = (const float*)d_in[14];

  const int N = in_sizes[0] / HDIM;
  const int E = in_sizes[2] / HDIM;
  const int L = in_sizes[4];
  const int G = out_size / OUTC - N;
  const int* src = eidx;
  const int* dst = eidx + E;

  // ---- workspace carve-up ----
  char* ws = (char*)d_ws;
  size_t off = 0;
  auto alloc = [&](size_t bytes) { char* p = ws + off; off += (bytes + 255) & ~size_t(255); return p; };
  float*    xf   = (float*)alloc((size_t)N * HDIM * 4);
  float*    hf   = (float*)alloc((size_t)N * HDIM * 4);
  float*    numf = (float*)alloc((size_t)N * HDIM * 4);
  float*    denf = (float*)alloc((size_t)N * HDIM * 4);
  unsigned* smax = (unsigned*)alloc((size_t)N * HDIM * 4);
  _Float16* a16  = (_Float16*)alloc((size_t)N * HDIM * 2);
  _Float16* m16  = (_Float16*)alloc((size_t)N * H2DIM * 2);
  _Float16* W1p  = (_Float16*)alloc((size_t)L * HDIM * H2DIM * 2);
  _Float16* W2p  = (_Float16*)alloc((size_t)L * H2DIM * HDIM * 2);
  _Float16* lWp  = (_Float16*)alloc((size_t)HDIM * OUTC * 2);
  float*    gsum = (float*)alloc((size_t)G * HDIM * 4);
  float*    gcnt = (float*)alloc((size_t)G * 4);
  (void)ws_size;

  const int NH = N * HDIM;
  const int bEdge = (E * 16 + 255) / 256;
  const int bNH   = (NH + 255) / 256;
  const int mtiles = N / 16;           // N = 50000 -> 3125 exact
  const int WSZ = HDIM * H2DIM;        // 8192 elems per layer weight

  // pack weights into WMMA-B fragment order (f16), once per launch
  for (int l = 0; l < L; ++l) {
    k_pack_b<<<(WSZ + 255) / 256, 256, 0, stream>>>(W1 + l * WSZ, W1p + l * WSZ, HDIM, H2DIM);
    k_pack_b<<<(WSZ + 255) / 256, 256, 0, stream>>>(W2 + l * WSZ, W2p + l * WSZ, H2DIM, HDIM);
  }
  k_pack_b<<<(WSZ + 255) / 256, 256, 0, stream>>>(linW, lWp, HDIM, OUTC);

  for (int l = 0; l < L; ++l) {
    const float* xin;
    if (l == 0) {
      xin = x;
    } else {
      k_prenorm_mish<<<(N + 7) / 8, 256, 0, stream>>>(xf, ng + l * HDIM, nb + l * HDIM,
                                                      hf, (_Float16*)nullptr, 0, N);
      xin = hf;
    }
    k_fill_u32<<<bNH, 256, 0, stream>>>(smax, ENC_NEGINF, NH);
    k_fill_u32<<<bNH, 256, 0, stream>>>((unsigned*)numf, 0u, NH);
    k_fill_u32<<<bNH, 256, 0, stream>>>((unsigned*)denf, 0u, NH);
    k_edge_max<<<bEdge, 256, 0, stream>>>(xin, ea, src, dst, t, l, smax, E);
    k_edge_accum<<<bEdge, 256, 0, stream>>>(xin, ea, src, dst, t, l, smax, numf, denf, E);
    k_combine<<<(NH / 4 + 255) / 256, 256, 0, stream>>>(numf, denf, xin, a16, NH / 4);
    k_gemm1<<<mtiles, 256, 0, stream>>>(a16, W1p + l * WSZ, b1 + l * H2DIM,
                                        lng + l * H2DIM, lnb + l * H2DIM, m16);
    k_gemm2<<<mtiles, 128, 0, stream>>>(m16, W2p + l * WSZ, b2 + l * HDIM,
                                        xf, (l == 0) ? 0 : 1);
  }

  // final prenorm with norm[0]; write f32 (pooling) and f16 (final GEMM)
  k_prenorm_mish<<<(N + 7) / 8, 256, 0, stream>>>(xf, ng, nb, hf, a16, 1, N);

  k_fill_u32<<<(G * HDIM + 255) / 256, 256, 0, stream>>>((unsigned*)gsum, 0u, G * HDIM);
  k_fill_u32<<<(G + 255) / 256, 256, 0, stream>>>((unsigned*)gcnt, 0u, G);
  k_pool<<<(N * 16 + 255) / 256, 256, 0, stream>>>(hf, batch, gsum, gcnt, N);

  float* out = (float*)d_out;
  k_gemm_out<<<mtiles, 256, 0, stream>>>(a16, lWp, linb, out);          // node outputs
  k_pool_out<<<G, OUTC, 0, stream>>>(gsum, gcnt, linW, linb, out + (size_t)N * OUTC);
}
